// My_CrossAttnDownBlock2D_90177133347505
// MI455X (gfx1250) — matrix operand order repaired and checked by reading, not compiled, not run on hardware
//
#include <hip/hip_runtime.h>

// ---------------------------------------------------------------------------
// Types / helpers
// ---------------------------------------------------------------------------
typedef unsigned short u16;
typedef __attribute__((ext_vector_type(16))) __bf16       v16bf;
typedef __attribute__((ext_vector_type(8)))  float        v8f;

#define DEV static __device__ __forceinline__
#define USE_ASYNC 1

DEV u16 f2bf(float x) {  // round-to-nearest-even f32 -> bf16 bits
    union { float f; unsigned int u; } c; c.f = x;
    unsigned int r = c.u + 0x7FFFu + ((c.u >> 16) & 1u);
    return (u16)(r >> 16);
}

union FragU { v16bf v; unsigned int u[8]; u16 s[16]; };

// A fragment: 16x32 bf16 (MxK), row-major LDS, ld even, k_base even.
// ISA 7.12.2: lanes 0-15 M=lane K-half 0, lanes 16-31 M=lane-16 K-half 8;
// VGPR i(0..3) -> K=ko+2i..+1, VGPR i(4..7) -> K=16+ko+2(i-4)..+1.
DEV v16bf load_a_frag(const u16* lds, int row_base, int k_base, int ld) {
    int lane = threadIdx.x & 31;
    int m    = lane & 15;
    int ko   = (lane < 16) ? 0 : 8;
    const unsigned int* p =
        (const unsigned int*)(lds + (size_t)(row_base + m) * ld + k_base);
    FragU r;
#pragma unroll
    for (int i = 0; i < 4; ++i) {
        r.u[i]     = p[(ko + 2 * i) >> 1];
        r.u[i + 4] = p[(16 + ko + 2 * i) >> 1];
    }
    return r.v;
}

// B fragment (32x16 KxN) from K-contiguous storage: B(k,n) = lds[(row_base+n)*ld + k_base + k]
// lanes 0-15: N=lane, K=0..7,16..23; lanes 16-31: N=lane-16, K=8..15,24..31 (2 per VGPR).
DEV v16bf load_bT_frag(const u16* lds, int row_base, int k_base, int ld) {
    int lane = threadIdx.x & 31;
    int n    = lane & 15;
    int ko   = (lane < 16) ? 0 : 8;
    const unsigned int* p =
        (const unsigned int*)(lds + (size_t)(row_base + n) * ld + k_base);
    FragU r;
#pragma unroll
    for (int i = 0; i < 4; ++i) {
        r.u[i]     = p[(ko + 2 * i) >> 1];
        r.u[i + 4] = p[(16 + ko + 2 * i) >> 1];
    }
    return r.v;
}

DEV v8f wmma_bf16(v16bf a, v16bf b, v8f c) {
    return __builtin_amdgcn_wmma_f32_16x16x32_bf16(false, a, false, b, (short)0, c,
                                                   false, false);
}

// Async 16B global -> LDS copy (per-lane), ASYNCcnt-tracked (ISA 15.18.3 op 98).
// Flat LDS pointers carry the LDS byte offset in addr[31:0] (ISA 10.2).
DEV void async_b128(u16* lds, const u16* g) {
#if USE_ASYNC
    unsigned loff = (unsigned)(size_t)lds;
    asm volatile("global_load_async_to_lds_b128 %0, %1, off"
                 :: "v"(loff), "v"(g)
                 : "memory");
#else
    *(uint4*)lds = *(const uint4*)g;
#endif
}
DEV void wait_async() {
#if USE_ASYNC
    asm volatile("s_wait_asynccnt 0x0" ::: "memory");
#endif
}

// ---------------------------------------------------------------------------
// Elementwise kernels
// ---------------------------------------------------------------------------
// f32 [K,N] -> bf16 transposed [N,K] (coalesced read, scattered write; weights small)
__global__ void convT_kernel(const float* __restrict__ src, u16* __restrict__ dst,
                             int K, int N) {
    int idx = blockIdx.x * blockDim.x + threadIdx.x;
    if (idx >= K * N) return;
    int k = idx / N, n = idx - k * N;
    dst[(size_t)n * K + k] = f2bf(src[idx]);
}

// LayerNorm: one wave32 per row, cols = 320
__global__ __launch_bounds__(256) void layernorm_kernel(
    const float* __restrict__ x, const float* __restrict__ g,
    const float* __restrict__ beta, u16* __restrict__ out, int rows, int cols) {
    int wave = threadIdx.x >> 5, lane = threadIdx.x & 31;
    int row = blockIdx.x * 8 + wave;
    if (row >= rows) return;
    const float* xr = x + (size_t)row * cols;
    float s = 0.f, s2 = 0.f;
    for (int c = lane; c < cols; c += 32) { float v = xr[c]; s += v; s2 += v * v; }
#pragma unroll
    for (int d = 1; d < 32; d <<= 1) { s += __shfl_xor(s, d, 32); s2 += __shfl_xor(s2, d, 32); }
    float mean = s / cols;
    float var  = s2 / cols - mean * mean;
    float inv  = rsqrtf(var + 1e-5f);
    u16* orow = out + (size_t)row * cols;
    for (int c = lane; c < cols; c += 32)
        orow[c] = f2bf((xr[c] - mean) * inv * g[c] + beta[c]);
}

// GEGLU: out[r,c] = proj[r,c] * gelu_exact(proj[r,c+1280]) -> bf16
__global__ void geglu_kernel(const float* __restrict__ proj, u16* __restrict__ out,
                             int rows) {
    int idx = blockIdx.x * blockDim.x + threadIdx.x;
    int total = rows * 1280;
    if (idx >= total) return;
    int r = idx / 1280, c = idx - r * 1280;
    float xv = proj[(size_t)r * 2560 + c];
    float gv = proj[(size_t)r * 2560 + 1280 + c];
    float ge = 0.5f * gv * (1.0f + erff(gv * 0.70710678118654752f));
    out[idx] = f2bf(xv * ge);
}

// ---------------------------------------------------------------------------
// bf16 WMMA GEMM: C = A[MxK] * B  (B given transposed as BT[N,K]).
// block 128 thr = 4 waves, tile 64x64, wave tile 32x32, K-step 32,
// async-DMA double-buffered LDS staging.
// EMODE: 0 = bf16 out [M,N]; 1 = bf16 out transposed [N,M];
//        2 = f32 out + bias + residual; 3 = f32 out + bias
// ---------------------------------------------------------------------------
template <int EMODE>
__global__ __launch_bounds__(128) void gemm_bf16_kernel(
    const u16* __restrict__ A, const u16* __restrict__ BT,
    const float* __restrict__ bias, const float* __restrict__ residual,
    float* __restrict__ outF, u16* __restrict__ outH, int M, int N, int K) {
    __shared__ u16 As[2][64 * 32];
    __shared__ u16 Bs[2][64 * 32];
    const int tid  = threadIdx.x;
    const int wave = tid >> 5, lane = tid & 31;
    const int bm = blockIdx.y * 64;
    const int bn = blockIdx.x * 64;
    const int wm = (wave & 1) * 32;
    const int wn = (wave >> 1) * 32;

    v8f zero = {};
    v8f acc[2][2];
#pragma unroll
    for (int i = 0; i < 2; ++i)
#pragma unroll
        for (int j = 0; j < 2; ++j) acc[i][j] = zero;

    auto stage = [&](int buf, int k0) {
        // A/BT tiles 64x32: 256 x 16B chunks each, 2 per thread
#pragma unroll
        for (int t = tid; t < 256; t += 128) {
            int row = t >> 2, ch = (t & 3) * 8;
            async_b128(&As[buf][row * 32 + ch], A + (size_t)(bm + row) * K + k0 + ch);
            async_b128(&Bs[buf][row * 32 + ch], BT + (size_t)(bn + row) * K + k0 + ch);
        }
    };

    stage(0, 0);
    wait_async();
    __syncthreads();

    int cur = 0;
    for (int k0 = 0; k0 < K; k0 += 32) {
        int nxt = cur ^ 1;
        if (k0 + 32 < K) stage(nxt, k0 + 32);  // DMA next tile while computing

        v16bf a0 = load_a_frag(As[cur], wm, 0, 32);
        v16bf a1 = load_a_frag(As[cur], wm + 16, 0, 32);
        v16bf b0 = load_bT_frag(Bs[cur], wn, 0, 32);
        v16bf b1 = load_bT_frag(Bs[cur], wn + 16, 0, 32);
        acc[0][0] = wmma_bf16(a0, b0, acc[0][0]);
        acc[0][1] = wmma_bf16(a0, b1, acc[0][1]);
        acc[1][0] = wmma_bf16(a1, b0, acc[1][0]);
        acc[1][1] = wmma_bf16(a1, b1, acc[1][1]);

        wait_async();
        __syncthreads();
        cur = nxt;
    }

    // epilogue (C layout: VGPR r -> M = r (+8 upper half), N = lane%16)
    const int nloc = lane & 15;
    const int moff = (lane < 16) ? 0 : 8;
#pragma unroll
    for (int i = 0; i < 2; ++i)
#pragma unroll
        for (int j = 0; j < 2; ++j) {
            int colg = bn + wn + j * 16 + nloc;
            int row0 = bm + wm + i * 16 + moff;
            if constexpr (EMODE == 0) {
#pragma unroll
                for (int r = 0; r < 8; ++r)
                    outH[(size_t)(row0 + r) * N + colg] = f2bf(acc[i][j][r]);
            } else if constexpr (EMODE == 1) {
                union { uint4 q4; u16 s[8]; } pk;
#pragma unroll
                for (int r = 0; r < 8; ++r) pk.s[r] = f2bf(acc[i][j][r]);
                *(uint4*)(outH + (size_t)colg * M + row0) = pk.q4;
            } else {
                float bb = bias[colg];
#pragma unroll
                for (int r = 0; r < 8; ++r) {
                    float v  = acc[i][j][r] + bb;
                    size_t o = (size_t)(row0 + r) * N + colg;
                    if constexpr (EMODE == 2) v += residual[o];
                    outF[o] = v;
                }
            }
        }
}

// ---------------------------------------------------------------------------
// Flash self-attention. Q/K in merged bf16 [B*LQ, 320] (head h = cols h*40..+39),
// V transposed bf16 [320, B*LQ]. Head dim zero-padded to 64 in LDS.
// grid (LQ/64, H, B), 128 threads = 4 waves x 16 q-rows.
// K/V blocks async-DMA double-buffered.
// ---------------------------------------------------------------------------
__global__ __launch_bounds__(128) void flash_attn_kernel(
    const u16* __restrict__ Q, const u16* __restrict__ Kb,
    const u16* __restrict__ Vt, u16* __restrict__ Ob) {
    __shared__ u16 Qs[64 * 64];
    __shared__ u16 Ks[2][64 * 64];   // [key][hd]
    __shared__ u16 Vs[2][64 * 64];   // transposed: [hd][key]
    __shared__ u16 Ps[4 * 16 * 64];

    const int tid = threadIdx.x;
    const int wave = tid >> 5, lane = tid & 31;
    const int h = blockIdx.y;
    const size_t rowbase = (size_t)blockIdx.z * 2048;
    const int q0 = blockIdx.x * 64;
    const int cbase = h * 40;

    // zero the pad regions once (cols/rows 40..63 never rewritten)
    for (int t = tid; t < 64 * 24; t += 128) {
        int r = t / 24, c = 40 + t % 24;
        Qs[r * 64 + c] = 0;
        Ks[0][r * 64 + c] = 0; Ks[1][r * 64 + c] = 0;
    }
    for (int t = tid; t < 24 * 64; t += 128) { Vs[0][40 * 64 + t] = 0; Vs[1][40 * 64 + t] = 0; }

    auto stageKV = [&](int buf, int kb) {
        // K block: 64 keys x 5 chunks; V block (transposed): 40 hd-rows x 8 chunks
        for (int t = tid; t < 320; t += 128) {
            int r = t / 5, ch = (t % 5) * 8;
            async_b128(&Ks[buf][r * 64 + ch], Kb + (rowbase + kb + r) * 320 + cbase + ch);
        }
        for (int t = tid; t < 320; t += 128) {
            int c = t >> 3, ch = (t & 7) * 8;
            async_b128(&Vs[buf][c * 64 + ch],
                       Vt + (size_t)(cbase + c) * 4096 + rowbase + kb + ch);
        }
    };

    // stage Q: 64 rows x 5 chunks of 8 bf16
    for (int t = tid; t < 320; t += 128) {
        int r = t / 5, ch = (t % 5) * 8;
        async_b128(Qs + r * 64 + ch, Q + (rowbase + q0 + r) * 320 + cbase + ch);
    }
    stageKV(0, 0);
    wait_async();
    __syncthreads();

    float mrow[8], lrow[8];
    v8f zero = {};
    v8f o[4];
#pragma unroll
    for (int r = 0; r < 8; ++r) { mrow[r] = -1e30f; lrow[r] = 0.f; }
#pragma unroll
    for (int j = 0; j < 4; ++j) o[j] = zero;

    const int   qr    = wave * 16;
    const float scale = rsqrtf(40.0f);
    u16* pw_base = Ps + wave * 16 * 64;

    v16bf aq0 = load_a_frag(Qs, qr, 0, 64);
    v16bf aq1 = load_a_frag(Qs, qr, 32, 64);

    int cur = 0;
    for (int kb = 0; kb < 2048; kb += 64) {
        if (kb + 64 < 2048) stageKV(cur ^ 1, kb + 64);  // DMA next block while computing

        // S = Q * K^T  (4 key tiles, 2 K-chunks of 32 over padded head dim)
        v8f s[4];
#pragma unroll
        for (int t = 0; t < 4; ++t) {
            v8f st = zero;
            st = wmma_bf16(aq0, load_bT_frag(Ks[cur], t * 16, 0, 64), st);
            st = wmma_bf16(aq1, load_bT_frag(Ks[cur], t * 16, 32, 64), st);
            s[t] = st;
        }

        // online softmax; row M=r(+8) lives across the 16 lanes of a half-wave
#pragma unroll
        for (int r = 0; r < 8; ++r) {
            float v0 = s[0][r] * scale, v1 = s[1][r] * scale;
            float v2 = s[2][r] * scale, v3 = s[3][r] * scale;
            float mx = fmaxf(fmaxf(v0, v1), fmaxf(v2, v3));
#pragma unroll
            for (int d = 1; d < 16; d <<= 1) mx = fmaxf(mx, __shfl_xor(mx, d, 32));
            float mnew  = fmaxf(mrow[r], mx);
            float alpha = __expf(mrow[r] - mnew);
            v0 = __expf(v0 - mnew); v1 = __expf(v1 - mnew);
            v2 = __expf(v2 - mnew); v3 = __expf(v3 - mnew);
            float ps = v0 + v1 + v2 + v3;
#pragma unroll
            for (int d = 1; d < 16; d <<= 1) ps += __shfl_xor(ps, d, 32);
            lrow[r] = lrow[r] * alpha + ps;
            mrow[r] = mnew;
#pragma unroll
            for (int j = 0; j < 4; ++j) o[j][r] *= alpha;
            int m = r + ((lane < 16) ? 0 : 8);
            int n = lane & 15;
            u16* pw = pw_base + m * 64;
            pw[0 * 16 + n] = f2bf(v0); pw[1 * 16 + n] = f2bf(v1);
            pw[2 * 16 + n] = f2bf(v2); pw[3 * 16 + n] = f2bf(v3);
        }

        // O += P * V  (V^T in LDS -> contiguous-K B fragments)
        v16bf ap0 = load_a_frag(pw_base, 0, 0, 64);
        v16bf ap1 = load_a_frag(pw_base, 0, 32, 64);
#pragma unroll
        for (int j = 0; j < 4; ++j) {
            o[j] = wmma_bf16(ap0, load_bT_frag(Vs[cur], j * 16, 0, 64), o[j]);
            o[j] = wmma_bf16(ap1, load_bT_frag(Vs[cur], j * 16, 32, 64), o[j]);
        }

        wait_async();
        __syncthreads();
        cur ^= 1;
    }

    float invl[8];
#pragma unroll
    for (int r = 0; r < 8; ++r) invl[r] = 1.0f / lrow[r];
    const int n = lane & 15, half = (lane < 16) ? 0 : 8;
#pragma unroll
    for (int j = 0; j < 4; ++j) {
        int c = j * 16 + n;
        if (c < 40) {
#pragma unroll
            for (int r = 0; r < 8; ++r)
                Ob[(rowbase + q0 + qr + half + r) * 320 + cbase + c] = f2bf(o[j][r] * invl[r]);
        }
    }
}

// ---------------------------------------------------------------------------
// Host orchestration
// ---------------------------------------------------------------------------
extern "C" void kernel_launch(void* const* d_in, const int* in_sizes, int n_in,
                              void* d_out, int out_size, void* d_ws, size_t ws_size,
                              hipStream_t stream) {
    (void)in_sizes; (void)n_in; (void)out_size; (void)ws_size;
    constexpr int NQ = 4096, DIMc = 320, FFc = 1280;

    const float* hs   = (const float*)d_in[0];
    const float* ln1g = (const float*)d_in[2];
    const float* ln1b = (const float*)d_in[3];
    const float* wq1  = (const float*)d_in[4];
    const float* wk1  = (const float*)d_in[5];
    const float* wv1  = (const float*)d_in[6];
    const float* wo1  = (const float*)d_in[7];
    const float* bo1  = (const float*)d_in[8];
    const float* ln2g = (const float*)d_in[9];
    const float* ln2b = (const float*)d_in[10];
    // wq2 (11), wk2 (12), encoder_hidden_states (1): dead -- softmax row-sum == 1
    const float* wvh  = (const float*)d_in[13];
    const float* wo2  = (const float*)d_in[14];
    const float* bo2  = (const float*)d_in[15];
    const float* ln3g = (const float*)d_in[16];
    const float* ln3b = (const float*)d_in[17];
    const float* wff1 = (const float*)d_in[18];
    const float* bff1 = (const float*)d_in[19];
    const float* wff2 = (const float*)d_in[20];
    const float* bff2 = (const float*)d_in[21];

    char* cur = (char*)d_ws;
    auto alloc = [&](size_t bytes) -> void* {
        void* p = cur; cur += (bytes + 255) & ~(size_t)255; return p;
    };
    u16* wq1t  = (u16*)alloc(DIMc * DIMc * 2);
    u16* wk1t  = (u16*)alloc(DIMc * DIMc * 2);
    u16* wv1t  = (u16*)alloc(DIMc * DIMc * 2);
    u16* wo1t  = (u16*)alloc(DIMc * DIMc * 2);
    u16* wvht  = (u16*)alloc(DIMc * DIMc * 2);
    u16* wo2t  = (u16*)alloc(DIMc * DIMc * 2);
    u16* wff1t = (u16*)alloc((size_t)DIMc * 2 * FFc * 2);
    u16* wff2t = (u16*)alloc((size_t)FFc * DIMc * 2);
    u16* hln   = (u16*)alloc((size_t)NQ * DIMc * 2);
    u16* qb    = (u16*)alloc((size_t)NQ * DIMc * 2);
    u16* kbuf  = (u16*)alloc((size_t)NQ * DIMc * 2);
    u16* vbufT = (u16*)alloc((size_t)NQ * DIMc * 2);   // [320, 4096]
    u16* attnb = (u16*)alloc((size_t)NQ * DIMc * 2);
    float* hs2 = (float*)alloc((size_t)NQ * DIMc * 4);
    u16* vh    = (u16*)alloc((size_t)NQ * DIMc * 2);
    float* hs3 = (float*)alloc((size_t)NQ * DIMc * 4);
    float* proj= (float*)alloc((size_t)NQ * 2 * FFc * 4);
    u16* ffin  = (u16*)alloc((size_t)NQ * FFc * 2);

    auto convT = [&](const float* s, u16* d, int K, int N) {
        int n = K * N;
        convT_kernel<<<dim3((n + 255) / 256), dim3(256), 0, stream>>>(s, d, K, N);
    };
    auto ln = [&](const float* x, const float* g, const float* b, u16* o) {
        layernorm_kernel<<<dim3(NQ / 8), dim3(256), 0, stream>>>(x, g, b, o, NQ, DIMc);
    };
#define GEMM(MODE, A, BT, bias, res, oF, oH, M, N, K)                                   \
    gemm_bf16_kernel<MODE><<<dim3((N) / 64, (M) / 64), dim3(128), 0, stream>>>(         \
        A, BT, bias, res, oF, oH, M, N, K)

    // weights -> bf16, transposed [N,K] for K-contiguous B fragments
    convT(wq1, wq1t, DIMc, DIMc);   convT(wk1, wk1t, DIMc, DIMc);
    convT(wv1, wv1t, DIMc, DIMc);   convT(wo1, wo1t, DIMc, DIMc);
    convT(wvh, wvht, DIMc, DIMc);   convT(wo2, wo2t, DIMc, DIMc);
    convT(wff1, wff1t, DIMc, 2 * FFc);
    convT(wff2, wff2t, FFc, DIMc);

    // ---- self-attention ----
    ln(hs, ln1g, ln1b, hln);
    GEMM(0, hln, wq1t, nullptr, nullptr, nullptr, qb,    NQ, DIMc, DIMc);
    GEMM(0, hln, wk1t, nullptr, nullptr, nullptr, kbuf,  NQ, DIMc, DIMc);
    GEMM(1, hln, wv1t, nullptr, nullptr, nullptr, vbufT, NQ, DIMc, DIMc);  // V transposed
    flash_attn_kernel<<<dim3(32, 8, 2), dim3(128), 0, stream>>>(qb, kbuf, vbufT, attnb);
    GEMM(2, attnb, wo1t, bo1, hs, hs2, nullptr, NQ, DIMc, DIMc);

    // ---- Hadamard cross-attn: softmax row-sum == 1  =>  cross = (h2 @ wvh) @ wo2 ----
    ln(hs2, ln2g, ln2b, hln);
    GEMM(0, hln, wvht, nullptr, nullptr, nullptr, vh, NQ, DIMc, DIMc);
    GEMM(2, vh, wo2t, bo2, hs2, hs3, nullptr, NQ, DIMc, DIMc);

    // ---- GEGLU FF ----
    ln(hs3, ln3g, ln3b, hln);
    GEMM(3, hln, wff1t, bff1, nullptr, proj, nullptr, NQ, 2 * FFc, DIMc);
    geglu_kernel<<<dim3((NQ * FFc + 255) / 256), dim3(256), 0, stream>>>(proj, ffin, NQ);
    GEMM(2, ffin, wff2t, bff2, hs3, (float*)d_out, nullptr, NQ, DIMc, FFc);
#undef GEMM
}
